// KidneyEdgePredictor_44650480009513
// MI455X (gfx1250) — compile-verified, matrix operand order checked by reading, not run on hardware
//
#include <hip/hip_runtime.h>

typedef __attribute__((ext_vector_type(16))) _Float16 v16h;
typedef __attribute__((ext_vector_type(8)))  _Float16 v8h;
typedef __attribute__((ext_vector_type(8)))  float    v8f;

static constexpr int   NNODES = 50000;
static constexpr int   NEDGES = 1280000;
static constexpr float SLOPE  = 0.2f;

// ---------------------------------------------------------------------------
// A-fragment loader: 16-bit A 16x32 layout (ISA 7.12.2). Lane l holds row
// M = l%16; its 16 f16 are K = {j0..j0+7} U {j0+16..j0+23} with j0 = hi?8:0.
// From a row-major f16 row this is two contiguous 16B loads.
// ---------------------------------------------------------------------------
static __device__ __forceinline__ v16h load_a_row(const _Float16* row, int kc, int hi) {
    const _Float16* p = row + kc * 32 + (hi ? 8 : 0);
    union { v16h v; v8h h[2]; } u;
    u.h[0] = *(const v8h*)(p);
    u.h[1] = *(const v8h*)(p + 16);
    return u.v;
}

static __device__ __forceinline__ v8f wmma_f16(v16h a, v16h b, v8f c) {
    return __builtin_amdgcn_wmma_f32_16x16x32_f16(false, a, false, b, (short)0, c, false, false);
}

// leaky_relu(x) == max(x, 0.2*x) since 0 < slope < 1 (v_mul + v_max, no VCC)
static __device__ __forceinline__ float leaky(float x) {
    return fmaxf(x, SLOPE * x);
}

// ---------------------------------------------------------------------------
// Pre-swizzle a row-major f32 weight [K x Ntot] into B-fragment-linear f16:
// frag[((kc*NT+nt)*32 + lane)*16 + j], K index mirrors the A layout transposed
// (lane = N within tile, j enumerates K runs). K >= Kvalid is zero-padded.
// ---------------------------------------------------------------------------
__global__ void prep_frag_kernel(const float* __restrict__ W, _Float16* __restrict__ out,
                                 int KC, int NT, int Ntot, int Kvalid) {
    int idx = blockIdx.x * blockDim.x + threadIdx.x;
    int total = KC * NT * 512;
    if (idx >= total) return;
    int j    = idx & 15;
    int lane = (idx >> 4) & 31;
    int t    = idx >> 9;          // kc*NT + nt
    int kc   = t / NT;
    int nt   = t - kc * NT;
    int kin  = (j & 7) + ((j & 8) ? 16 : 0) + ((lane & 16) ? 8 : 0);
    int K    = kc * 32 + kin;
    int N    = nt * 16 + (lane & 15);
    float v  = (K < Kvalid) ? W[(unsigned)K * (unsigned)Ntot + (unsigned)N] : 0.f;
    out[(unsigned)idx] = (_Float16)v;
}

// edge_attr = [x[src](13) | x[dst](13) | raw(1) | 0-pad(5)] as f16 [E x 32];
// degree counts fused (same per-edge pass over src/dst).
__global__ void build_ea_kernel(const float* __restrict__ x, const int* __restrict__ src,
                                const int* __restrict__ dst, const float* __restrict__ raw,
                                _Float16* __restrict__ ea,
                                float* __restrict__ cnt_src, float* __restrict__ cnt_dst) {
    unsigned e = blockIdx.x * blockDim.x + threadIdx.x;
    if (e >= (unsigned)NEDGES) return;
    unsigned sn = (unsigned)src[e], dn = (unsigned)dst[e];
    atomicAdd(cnt_src + sn, 1.f);
    atomicAdd(cnt_dst + dn, 1.f);
    const float* xs = x + sn * 13u;
    const float* xd = x + dn * 13u;
    _Float16 row[32];
#pragma unroll
    for (int i = 0; i < 13; ++i) { row[i] = (_Float16)xs[i]; row[13 + i] = (_Float16)xd[i]; }
    row[26] = (_Float16)raw[e];
#pragma unroll
    for (int i = 27; i < 32; ++i) row[i] = (_Float16)0.f;
    v8h* o = (v8h*)(ea + e * 32u);
    const v8h* r = (const v8h*)row;
    o[0] = r[0]; o[1] = r[1]; o[2] = r[2]; o[3] = r[3];
}

// h1 = leaky(edge_attr @ enc_w + enc_b); fused scatter-add into node sums.
__global__ __launch_bounds__(128) void encoder_kernel(
    const _Float16* __restrict__ ea, const _Float16* __restrict__ wfrag,
    const float* __restrict__ bias, const int* __restrict__ src, const int* __restrict__ dst,
    _Float16* __restrict__ hout, float* __restrict__ Sdst, float* __restrict__ Ssrc) {
    unsigned lane = threadIdx.x & 31u;
    unsigned wave = threadIdx.x >> 5;
    unsigned ebase = (blockIdx.x * 4u + wave) * 16u;
    unsigned lm = lane & 15u, hi = lane >> 4;

    v8f acc[4];
#pragma unroll
    for (int nt = 0; nt < 4; ++nt) {
        float bv = bias[nt * 16u + lm];
#pragma unroll
        for (int i = 0; i < 8; ++i) acc[nt][i] = bv;
    }
    v16h a = load_a_row(ea + (ebase + lm) * 32u, 0, (int)hi);
#pragma unroll
    for (int nt = 0; nt < 4; ++nt) {
        v16h b = *(const v16h*)(wfrag + ((unsigned)nt * 32u + lane) * 16u);
        acc[nt] = wmma_f16(a, b, acc[nt]);
    }
#pragma unroll
    for (unsigned r = 0; r < 8; ++r) {
        unsigned e = ebase + r + 8u * hi;
        unsigned sn = (unsigned)src[e], dn = (unsigned)dst[e];
        unsigned ho = e * 64u + lm;
        unsigned od = dn * 64u + lm;
        unsigned os = sn * 64u + lm;
#pragma unroll
        for (unsigned nt = 0; nt < 4; ++nt) {
            float v = leaky(acc[nt][r]);
            hout[ho + nt * 16u] = (_Float16)v;
            atomicAdd(Sdst + od + nt * 16u, v);
            atomicAdd(Ssrc + os + nt * 16u, v);
        }
    }
}

// Tbl = f16(S / max(cnt,1)); reset S for next layer's scatter.
__global__ void normalize_kernel(float* __restrict__ Sdst, float* __restrict__ Ssrc,
                                 const float* __restrict__ cnt_dst, const float* __restrict__ cnt_src,
                                 _Float16* __restrict__ Tdst, _Float16* __restrict__ Tsrc) {
    unsigned i = blockIdx.x * blockDim.x + threadIdx.x;
    if (i >= (unsigned)(NNODES * 64)) return;
    unsigned n = i >> 6;
    Tdst[i] = (_Float16)(Sdst[i] / fmaxf(cnt_dst[n], 1.f));
    Tsrc[i] = (_Float16)(Ssrc[i] / fmaxf(cnt_src[n], 1.f));
    Sdst[i] = 0.f;
    Ssrc[i] = 0.f;
}

// h' = leaky(h@Wself + b + Tdst[src]@Win + Tsrc[dst]@Wout); optional scatter.
__global__ __launch_bounds__(128) void conv_kernel(
    const _Float16* __restrict__ hin, const _Float16* __restrict__ Tdst,
    const _Float16* __restrict__ Tsrc, const _Float16* __restrict__ wself,
    const _Float16* __restrict__ win, const _Float16* __restrict__ wout,
    const float* __restrict__ bias, const int* __restrict__ src, const int* __restrict__ dst,
    _Float16* __restrict__ hout, float* __restrict__ Sdst, float* __restrict__ Ssrc,
    int do_scatter) {
    unsigned lane = threadIdx.x & 31u;
    unsigned wave = threadIdx.x >> 5;
    unsigned ebase = (blockIdx.x * 4u + wave) * 16u;
    unsigned lm = lane & 15u, hi = lane >> 4;

    unsigned esel = ebase + lm;
    const _Float16* rows[3];
    rows[0] = hin + esel * 64u;
    rows[1] = Tdst + (unsigned)src[esel] * 64u;   // agg_in  = meanByDst[src[e]]
    rows[2] = Tsrc + (unsigned)dst[esel] * 64u;   // agg_out = meanBySrc[dst[e]]
    const _Float16* ws[3] = { wself, win, wout };

    v8f acc[4];
#pragma unroll
    for (int nt = 0; nt < 4; ++nt) {
        float bv = bias[nt * 16u + lm];
#pragma unroll
        for (int i = 0; i < 8; ++i) acc[nt][i] = bv;
    }
#pragma unroll
    for (int g = 0; g < 3; ++g) {
#pragma unroll
        for (int kc = 0; kc < 2; ++kc) {
            v16h a = load_a_row(rows[g], kc, (int)hi);
#pragma unroll
            for (int nt = 0; nt < 4; ++nt) {
                v16h b = *(const v16h*)(ws[g] + ((unsigned)(kc * 4 + nt) * 32u + lane) * 16u);
                acc[nt] = wmma_f16(a, b, acc[nt]);
            }
        }
    }
#pragma unroll
    for (unsigned r = 0; r < 8; ++r) {
        unsigned e = ebase + r + 8u * hi;
        unsigned sn = (unsigned)src[e], dn = (unsigned)dst[e];
        unsigned ho = e * 64u + lm;
        unsigned od = dn * 64u + lm;
        unsigned os = sn * 64u + lm;
#pragma unroll
        for (unsigned nt = 0; nt < 4; ++nt) {
            float v = leaky(acc[nt][r]);
            hout[ho + nt * 16u] = (_Float16)v;
            if (do_scatter) {
                atomicAdd(Sdst + od + nt * 16u, v);
                atomicAdd(Ssrc + os + nt * 16u, v);
            }
        }
    }
}

// MLP 64->64->64->32->1 fused: LDS round-trips convert C frags to A frags.
__global__ __launch_bounds__(128) void mlp_kernel(
    const _Float16* __restrict__ hin,
    const _Float16* __restrict__ w1f, const float* __restrict__ b1,
    const _Float16* __restrict__ w2f, const float* __restrict__ b2,
    const _Float16* __restrict__ w3f, const float* __restrict__ b3,
    const float* __restrict__ w4, const float* __restrict__ b4,
    float* __restrict__ out) {
    __shared__ __align__(16) _Float16 lds[4][16][72];
    unsigned lane = threadIdx.x & 31u;
    unsigned wave = threadIdx.x >> 5;
    unsigned ebase = (blockIdx.x * 4u + wave) * 16u;
    unsigned lm = lane & 15u, hi = lane >> 4;

    // ---- layer 1: hin @ w1 + b1, leaky -> LDS rows ----
    v8f acc[4];
#pragma unroll
    for (int nt = 0; nt < 4; ++nt) {
        float bv = b1[nt * 16u + lm];
#pragma unroll
        for (int i = 0; i < 8; ++i) acc[nt][i] = bv;
    }
#pragma unroll
    for (int kc = 0; kc < 2; ++kc) {
        v16h a = load_a_row(hin + (ebase + lm) * 64u, kc, (int)hi);
#pragma unroll
        for (int nt = 0; nt < 4; ++nt) {
            v16h b = *(const v16h*)(w1f + ((unsigned)(kc * 4 + nt) * 32u + lane) * 16u);
            acc[nt] = wmma_f16(a, b, acc[nt]);
        }
    }
#pragma unroll
    for (unsigned r = 0; r < 8; ++r)
#pragma unroll
        for (unsigned nt = 0; nt < 4; ++nt)
            lds[wave][r + 8u * hi][nt * 16u + lm] = (_Float16)leaky(acc[nt][r]);
    __syncthreads();

    // ---- layer 2 ----
    v16h a0 = load_a_row(&lds[wave][lm][0], 0, (int)hi);
    v16h a1 = load_a_row(&lds[wave][lm][0], 1, (int)hi);
    __syncthreads();
#pragma unroll
    for (int nt = 0; nt < 4; ++nt) {
        float bv = b2[nt * 16u + lm];
#pragma unroll
        for (int i = 0; i < 8; ++i) acc[nt][i] = bv;
        v16h bl = *(const v16h*)(w2f + ((unsigned)(0 * 4 + nt) * 32u + lane) * 16u);
        acc[nt] = wmma_f16(a0, bl, acc[nt]);
        v16h bh = *(const v16h*)(w2f + ((unsigned)(1 * 4 + nt) * 32u + lane) * 16u);
        acc[nt] = wmma_f16(a1, bh, acc[nt]);
    }
#pragma unroll
    for (unsigned r = 0; r < 8; ++r)
#pragma unroll
        for (unsigned nt = 0; nt < 4; ++nt)
            lds[wave][r + 8u * hi][nt * 16u + lm] = (_Float16)leaky(acc[nt][r]);
    __syncthreads();

    // ---- layer 3 (64 -> 32, NT=2) ----
    a0 = load_a_row(&lds[wave][lm][0], 0, (int)hi);
    a1 = load_a_row(&lds[wave][lm][0], 1, (int)hi);
    v8f acc3[2];
#pragma unroll
    for (int nt = 0; nt < 2; ++nt) {
        float bv = b3[nt * 16u + lm];
#pragma unroll
        for (int i = 0; i < 8; ++i) acc3[nt][i] = bv;
        v16h bl = *(const v16h*)(w3f + ((unsigned)(0 * 2 + nt) * 32u + lane) * 16u);
        acc3[nt] = wmma_f16(a0, bl, acc3[nt]);
        v16h bh = *(const v16h*)(w3f + ((unsigned)(1 * 2 + nt) * 32u + lane) * 16u);
        acc3[nt] = wmma_f16(a1, bh, acc3[nt]);
    }

    // ---- layer 4: (32 -> 1) dot + cross-lane reduce ----
    float w4a = w4[lm], w4b = w4[16u + lm];
    float bb = b4[0];
#pragma unroll
    for (unsigned r = 0; r < 8; ++r) {
        float p = leaky(acc3[0][r]) * w4a + leaky(acc3[1][r]) * w4b;
        p += __shfl_xor(p, 1, 32);
        p += __shfl_xor(p, 2, 32);
        p += __shfl_xor(p, 4, 32);
        p += __shfl_xor(p, 8, 32);
        if (lm == 0) out[ebase + r + 8u * hi] = p + bb;
    }
}

// ---------------------------------------------------------------------------
extern "C" void kernel_launch(void* const* d_in, const int* in_sizes, int n_in,
                              void* d_out, int out_size, void* d_ws, size_t ws_size,
                              hipStream_t stream) {
    (void)in_sizes; (void)n_in; (void)out_size; (void)ws_size;
    const float* x     = (const float*)d_in[0];
    const int*   ei    = (const int*)d_in[1];
    const int*   src   = ei;
    const int*   dst   = ei + NEDGES;
    const float* raw   = (const float*)d_in[2];
    const float* enc_w = (const float*)d_in[3];
    const float* enc_b = (const float*)d_in[4];
    const float* w_self[3] = { (const float*)d_in[5],  (const float*)d_in[9],  (const float*)d_in[13] };
    const float* b_self[3] = { (const float*)d_in[6],  (const float*)d_in[10], (const float*)d_in[14] };
    const float* w_in[3]   = { (const float*)d_in[7],  (const float*)d_in[11], (const float*)d_in[15] };
    const float* w_out[3]  = { (const float*)d_in[8],  (const float*)d_in[12], (const float*)d_in[16] };
    const float* mw1 = (const float*)d_in[17]; const float* mb1 = (const float*)d_in[18];
    const float* mw2 = (const float*)d_in[19]; const float* mb2 = (const float*)d_in[20];
    const float* mw3 = (const float*)d_in[21]; const float* mb3 = (const float*)d_in[22];
    const float* mw4 = (const float*)d_in[23]; const float* mb4 = (const float*)d_in[24];
    float* out = (float*)d_out;

    const size_t NND = (size_t)NNODES * 64;
    char* ws = (char*)d_ws;
    size_t off = 0;
    auto take = [&](size_t bytes) -> char* {
        char* p = ws + off;
        off = (off + bytes + 255) & ~(size_t)255;
        return p;
    };
    float*    Sdst    = (float*)take(NND * 4);
    float*    Ssrc    = (float*)take(NND * 4);
    float*    cnt_dst = (float*)take((size_t)NNODES * 4);
    float*    cnt_src = (float*)take((size_t)NNODES * 4);
    _Float16* Tdst    = (_Float16*)take(NND * 2);
    _Float16* Tsrc    = (_Float16*)take(NND * 2);
    _Float16* f_enc   = (_Float16*)take(2048 * 2);
    _Float16* f_self[3]; _Float16* f_in[3]; _Float16* f_out[3];
    for (int l = 0; l < 3; ++l) {
        f_self[l] = (_Float16*)take(4096 * 2);
        f_in[l]   = (_Float16*)take(4096 * 2);
        f_out[l]  = (_Float16*)take(4096 * 2);
    }
    _Float16* f_m1 = (_Float16*)take(4096 * 2);
    _Float16* f_m2 = (_Float16*)take(4096 * 2);
    _Float16* f_m3 = (_Float16*)take(2048 * 2);
    _Float16* HA   = (_Float16*)take((size_t)NEDGES * 64 * 2);
    _Float16* HB   = (_Float16*)take((size_t)NEDGES * 64 * 2);
    _Float16* EA   = HB;  // edge_attr aliases HB (dead before first write to HB)

    hipMemsetAsync(Sdst, 0, NND * 4, stream);
    hipMemsetAsync(Ssrc, 0, NND * 4, stream);
    hipMemsetAsync(cnt_dst, 0, (size_t)NNODES * 4, stream);
    hipMemsetAsync(cnt_src, 0, (size_t)NNODES * 4, stream);

    auto prep = [&](const float* W, _Float16* o, int KC, int NT, int Ntot, int Kvalid) {
        int total = KC * NT * 512;
        prep_frag_kernel<<<(total + 255) / 256, 256, 0, stream>>>(W, o, KC, NT, Ntot, Kvalid);
    };
    prep(enc_w, f_enc, 1, 4, 64, 27);
    for (int l = 0; l < 3; ++l) {
        prep(w_self[l], f_self[l], 2, 4, 64, 64);
        prep(w_in[l],   f_in[l],   2, 4, 64, 64);
        prep(w_out[l],  f_out[l],  2, 4, 64, 64);
    }
    prep(mw1, f_m1, 2, 4, 64, 64);
    prep(mw2, f_m2, 2, 4, 64, 64);
    prep(mw3, f_m3, 2, 2, 32, 64);

    build_ea_kernel<<<NEDGES / 256, 256, 0, stream>>>(x, src, dst, raw, EA, cnt_src, cnt_dst);

    const int GEMM_BLOCKS = NEDGES / 64;  // 4 waves/block, 16 edges/wave
    encoder_kernel<<<GEMM_BLOCKS, 128, 0, stream>>>(EA, f_enc, enc_b, src, dst, HA, Sdst, Ssrc);

    const int NORM_BLOCKS = (NNODES * 64 + 255) / 256;
    normalize_kernel<<<NORM_BLOCKS, 256, 0, stream>>>(Sdst, Ssrc, cnt_dst, cnt_src, Tdst, Tsrc);
    conv_kernel<<<GEMM_BLOCKS, 128, 0, stream>>>(HA, Tdst, Tsrc, f_self[0], f_in[0], f_out[0],
                                                 b_self[0], src, dst, HB, Sdst, Ssrc, 1);
    normalize_kernel<<<NORM_BLOCKS, 256, 0, stream>>>(Sdst, Ssrc, cnt_dst, cnt_src, Tdst, Tsrc);
    conv_kernel<<<GEMM_BLOCKS, 128, 0, stream>>>(HB, Tdst, Tsrc, f_self[1], f_in[1], f_out[1],
                                                 b_self[1], src, dst, HA, Sdst, Ssrc, 1);
    normalize_kernel<<<NORM_BLOCKS, 256, 0, stream>>>(Sdst, Ssrc, cnt_dst, cnt_src, Tdst, Tsrc);
    conv_kernel<<<GEMM_BLOCKS, 128, 0, stream>>>(HA, Tdst, Tsrc, f_self[2], f_in[2], f_out[2],
                                                 b_self[2], src, dst, HB, Sdst, Ssrc, 0);

    mlp_kernel<<<GEMM_BLOCKS, 128, 0, stream>>>(HB, f_m1, mb1, f_m2, mb2, f_m3, mb3, mw4, mb4, out);
}
